// AutoRegressiveLSTM_77017353552557
// MI455X (gfx1250) — matrix operand order repaired and checked by reading, not compiled, not run on hardware
//
#include <hip/hip_runtime.h>

typedef __attribute__((ext_vector_type(16))) __bf16 v16bf;
typedef __attribute__((ext_vector_type(8)))  __bf16 v8bf;
typedef __attribute__((ext_vector_type(8)))  float  v8f;

#define T_STEPS   96
#define OUT_STEPS 24
#define HSTR      72    // bf16 row stride of per-wave h buffer (64 cols + pad); 144B = 16B aligned
#define HBUF      (16 * HSTR)  // one h buffer (elems); two per wave for double buffering
#define CSTR      68    // f32 row stride of per-wave cell state; 8*68 mod 64 = 32 -> halves on disjoint banks
#define KC        2     // K chunks of 32 over K=64 (pure h @ U)
#define NT        16    // N tiles over the 256 gate columns
#define WAVES     8     // waves per workgroup
#define BPW       16    // batch rows per wave

static __device__ __forceinline__ float fast_sigmoid(float x) {
  float e = __expf(-x);
  return __builtin_amdgcn_rcpf(1.0f + e);
}
static __device__ __forceinline__ float fast_tanh(float x) {
#if __has_builtin(__builtin_amdgcn_tanhf)
  return __builtin_amdgcn_tanhf(x);
#else
  float e = __expf(-2.0f * x);
  return (1.0f - e) * __builtin_amdgcn_rcpf(1.0f + e);
#endif
}

__global__ __launch_bounds__(256) void lstm_ar_kernel(
    const float* __restrict__ inp,
    const float* __restrict__ Wg,
    const float* __restrict__ Ug,
    const float* __restrict__ bia,
    const float* __restrict__ Wd,
    const float* __restrict__ bd,
    float* __restrict__ out, int Bn) {
  extern __shared__ __align__(32) char smem[];
  __bf16* sFragB = (__bf16*)smem;                    // NT*KC*512 = 16384 elems (32KB): U B-fragments
  __bf16* sFragD = sFragB + NT * KC * 512;           // KC*512 = 1024 elems: padded Wd B-fragments
  float*  sWp    = (float*)(sFragD + KC * 512);      // 512 f32: W pairs, sWp[n*2+j] = W[j][n]
  float*  sBias  = sWp + 512;                        // 256 f32
  __bf16* sH     = (__bf16*)(sBias + 256);           // WAVES * 2 * HBUF bf16: per-wave h, double buffered
  float*  sC     = (float*)(sH + WAVES * 2 * HBUF);  // WAVES*16*CSTR f32: per-wave cell state
  float*  sX     = sC + WAVES * 16 * CSTR;           // WAVES*16*2 f32: per-wave x (kept in f32)

  const int tid = threadIdx.x;

  // ---- One-time pre-swizzle of U into WMMA B-fragment layout ----
  // B layout (16-bit, 32x16): lane = N col; element s = 2v+t holds K = 8*half + (s&7) + 16*(s>>3)
  // (table order, matching loadA). Flat short index:
  //   idx = ((jj*KC + kk)*32 + lane)*16 + 2v + t
  for (int idx = tid; idx < NT * KC * 512; idx += 256) {
    int t = idx & 1, v = (idx >> 1) & 7, l = (idx >> 4) & 31;
    int rest = idx >> 9; int kk = rest & 1, jj = rest >> 1;
    int n = jj * 16 + (l & 15);
    int k = kk * 32 + 2 * (v & 3) + t + 8 * (l >> 4) + 16 * (v >> 2);     // 0..63: row of U
    sFragB[idx] = (__bf16)Ug[k * 256 + n];
  }
  for (int idx = tid; idx < KC * 512; idx += 256) {
    int t = idx & 1, v = (idx >> 1) & 7, l = (idx >> 4) & 31, kk = idx >> 9;
    int n = l & 15;
    int k = kk * 32 + 2 * (v & 3) + t + 8 * (l >> 4) + 16 * (v >> 2);
    sFragD[idx] = (__bf16)((n < 2) ? Wd[k * 2 + n] : 0.0f);
  }
  for (int idx = tid; idx < 512; idx += 256) sWp[idx] = Wg[(idx & 1) * 256 + (idx >> 1)];
  sBias[tid] = bia[tid];
  for (int idx = tid; idx < WAVES * 2 * HBUF; idx += 256) sH[idx] = (__bf16)0.0f;
  for (int idx = tid; idx < WAVES * 16 * CSTR; idx += 256) sC[idx] = 0.0f;
  for (int idx = tid; idx < WAVES * 32; idx += 256) sX[idx] = 0.0f;
  __syncthreads();

  const int lane = tid & 31, wv = tid >> 5;
  const int lh = lane >> 4, ln = lane & 15;
  const int batch0 = (blockIdx.x * WAVES + wv) * BPW;
  const bool valid = (batch0 + BPW) <= Bn;
  const int bsafe = valid ? batch0 : (Bn >= BPW ? Bn - BPW : 0);
  __bf16* hb = sH + wv * 2 * HBUF;     // [2][16][HSTR] double-buffered h
  float*  cb = sC + wv * 16 * CSTR;
  float*  xb_ = sX + wv * 32;

  const float bdv = (ln < 2) ? bd[ln] : 0.0f;

  // A operand loader: 16-bit A 16x32 layout -> two contiguous 16B chunks per lane, per K-chunk.
  auto loadA = [&](const __bf16* hr, v16bf A[KC]) {
#pragma unroll
    for (int kk = 0; kk < KC; ++kk) {
      int base = kk * 32 + 8 * lh;
      v8bf lo = *(const v8bf*)(hr + ln * HSTR + base);
      v8bf hi = *(const v8bf*)(hr + ln * HSTR + base + 16);
      A[kk] = __builtin_shufflevector(lo, hi, 0, 1, 2, 3, 4, 5, 6, 7,
                                      8, 9, 10, 11, 12, 13, 14, 15);
    }
  };

  // One fused LSTM step: reads h from hr, writes new h to hw (double buffer -> no RAW hazard).
  // z = h@U (WMMA, inline-0 C) + x@W + b (epilogue); gate math; c in LDS.
  auto lstm_step = [&](const __bf16* hr, __bf16* hw) {
    float2 xv[8];
#pragma unroll
    for (int e = 0; e < 8; ++e) xv[e] = *(const float2*)(xb_ + (e + 8 * lh) * 2); // broadcast reads
    v16bf A[KC];
    loadA(hr, A);
#pragma unroll 1
    for (int q = 0; q < 4; ++q) {           // gate group: i=q, f=q+4, g=q+8, o=q+12
      float2 wI = *(const float2*)(sWp + (       q * 16 + ln) * 2);
      float2 wF = *(const float2*)(sWp + ( 64 + q * 16 + ln) * 2);
      float2 wG = *(const float2*)(sWp + (128 + q * 16 + ln) * 2);
      float2 wO = *(const float2*)(sWp + (192 + q * 16 + ln) * 2);
      float bi  = sBias[       q * 16 + ln];
      float bff = sBias[ 64 + q * 16 + ln];
      float bgg = sBias[128 + q * 16 + ln];
      float boo = sBias[192 + q * 16 + ln];
      v8f zi = {}, zf = {}, zg = {}, zo = {};   // zeroinit C -> WMMA inline-0 SRC2
#pragma unroll
      for (int kk = 0; kk < KC; ++kk) {
        v16bf Bi = *(const v16bf*)(sFragB + ((q     ) * KC + kk) * 512 + lane * 16);
        zi = __builtin_amdgcn_wmma_f32_16x16x32_bf16(false, A[kk], false, Bi, (short)0, zi, false, false);
        v16bf Bf = *(const v16bf*)(sFragB + ((q +  4) * KC + kk) * 512 + lane * 16);
        zf = __builtin_amdgcn_wmma_f32_16x16x32_bf16(false, A[kk], false, Bf, (short)0, zf, false, false);
        v16bf Bg = *(const v16bf*)(sFragB + ((q +  8) * KC + kk) * 512 + lane * 16);
        zg = __builtin_amdgcn_wmma_f32_16x16x32_bf16(false, A[kk], false, Bg, (short)0, zg, false, false);
        v16bf Bo = *(const v16bf*)(sFragB + ((q + 12) * KC + kk) * 512 + lane * 16);
        zo = __builtin_amdgcn_wmma_f32_16x16x32_bf16(false, A[kk], false, Bo, (short)0, zo, false, false);
      }
#pragma unroll
      for (int e = 0; e < 8; ++e) {
        int m = e + 8 * lh;
        float xa = xv[e].x, xbv = xv[e].y;
        float zie = zi[e] + bi  + xa * wI.x + xbv * wI.y;
        float zfe = zf[e] + bff + xa * wF.x + xbv * wF.y;
        float zge = zg[e] + bgg + xa * wG.x + xbv * wG.y;
        float zoe = zo[e] + boo + xa * wO.x + xbv * wO.y;
        float cold = cb[m * CSTR + q * 16 + ln];
        float cn = fast_sigmoid(zfe) * cold + fast_sigmoid(zie) * fast_tanh(zge);
        cb[m * CSTR + q * 16 + ln] = cn;
        float hn = fast_sigmoid(zoe) * fast_tanh(cn);
        hw[m * HSTR + q * 16 + ln] = (__bf16)hn;
      }
    }
  };

  // pred = h @ Wd + bd via WMMA against padded Wd fragments; writes output + feeds x buffer (f32).
  auto do_pred = [&](const __bf16* hr, int s) {
    v16bf A[KC];
    loadA(hr, A);
    v8f p = {};
#pragma unroll
    for (int kk = 0; kk < KC; ++kk) {
      v16bf Bp = *(const v16bf*)(sFragD + kk * 512 + lane * 16);
      p = __builtin_amdgcn_wmma_f32_16x16x32_bf16(false, A[kk], false, Bp, (short)0, p, false, false);
    }
    if (ln < 2) {
#pragma unroll
      for (int e = 0; e < 8; ++e) {
        int m = e + 8 * lh;
        float pv = p[e] + bdv;
        xb_[m * 2 + ln] = pv;                                    // feed back as next x (f32)
        if (valid)
          out[(size_t)(batch0 + m) * (OUT_STEPS * 2) + s * 2 + ln] = pv;
      }
    }
  };

  // ---- Warmup over the 96-step input sequence ----
  const float* xp = inp + (size_t)(bsafe + lane) * (T_STEPS * 2);
  float xa = 0.0f, xbv = 0.0f;
  if (lane < 16) { xa = xp[0]; xbv = xp[1]; }
  int p = 0;                                 // current h buffer index
#pragma unroll 1
  for (int t = 0; t < T_STEPS; ++t) {
    if (lane < 16) {
      xb_[lane * 2 + 0] = xa;
      xb_[lane * 2 + 1] = xbv;
      if (t + 1 < T_STEPS) { xa = xp[2 * t + 2]; xbv = xp[2 * t + 3]; }  // prefetch next step
    }
    lstm_step(hb + p * HBUF, hb + (p ^ 1) * HBUF);
    p ^= 1;
  }

  // ---- Autoregressive decode ----
  do_pred(hb + p * HBUF, 0);
#pragma unroll 1
  for (int s = 1; s < OUT_STEPS; ++s) {
    lstm_step(hb + p * HBUF, hb + (p ^ 1) * HBUF);
    p ^= 1;
    do_pred(hb + p * HBUF, s);
  }
}

extern "C" void kernel_launch(void* const* d_in, const int* in_sizes, int n_in,
                              void* d_out, int out_size, void* d_ws, size_t ws_size,
                              hipStream_t stream) {
  const float* inp = (const float*)d_in[0];
  const float* Wg  = (const float*)d_in[1];
  const float* Ug  = (const float*)d_in[2];
  const float* bia = (const float*)d_in[3];
  const float* Wd  = (const float*)d_in[4];
  const float* bd  = (const float*)d_in[5];
  float* out = (float*)d_out;

  int Bn = in_sizes[0] / (T_STEPS * 2);
  int grid = (Bn + WAVES * BPW - 1) / (WAVES * BPW);   // 512 workgroups for B=65536
  size_t shmem = (size_t)(NT * KC * 512 + KC * 512) * sizeof(short)  // U + Wd fragments
               + (512 + 256) * sizeof(float)                          // W pairs + bias
               + (size_t)WAVES * 2 * HBUF * sizeof(short)             // per-wave h, double buffered
               + (size_t)WAVES * 16 * CSTR * sizeof(float)            // per-wave cell state
               + (size_t)WAVES * 32 * sizeof(float);                  // per-wave x (f32)
  lstm_ar_kernel<<<grid, 256, shmem, stream>>>(inp, Wg, Ug, bia, Wd, bd, out, Bn);
}
